// Embeddings1DConnector_26989574488642
// MI455X (gfx1250) — compile-verified, matrix operand order checked by reading, not run on hardware
//
#include <hip/hip_runtime.h>
#include <hip/hip_bf16.h>
#include <math.h>

// ---------------------------------------------------------------------------
// Problem constants (match reference): B=4, S=1024, H=16, HD=64, D=1024, L=2
// ---------------------------------------------------------------------------
#define S_LEN   1024
#define NB      4
#define NH      16
#define HDIM    64
#define DMODEL  1024
#define NLAYER  2
#define NREG    128
#define MTOK    (NB * S_LEN)      /* 4096 rows of activations */

typedef __bf16 v16bf __attribute__((ext_vector_type(16)));
typedef float  v8f   __attribute__((ext_vector_type(8)));

#define WMMA_BF16(a, b, c) \
    __builtin_amdgcn_wmma_f32_16x16x32_bf16(false, (a), false, (b), (short)0, (c), false, false)

// float -> bf16 bits, round-to-nearest-even, NaN/Inf passthrough
__device__ __forceinline__ unsigned short f2bf(float f) {
    union { float f; unsigned u; } v; v.f = f;
    unsigned u = v.u;
    unsigned r = u + 0x7fffu + ((u >> 16) & 1u);
    if ((u & 0x7f800000u) == 0x7f800000u) r = u;
    return (unsigned short)(r >> 16);
}

// Fragment-buffer index for element (r, k) of a matrix with contracted extent
// Ktot. For an A operand r = M row; for a B operand r = N column (identical
// packing per the CDNA5 ISA 16-bit operand layout, 16x32 tiles, 512 elem/tile).
__device__ __forceinline__ unsigned fragIdx(unsigned r, unsigned k, unsigned Ktot) {
    unsigned kk       = k & 31u;
    unsigned lanehalf = (kk >> 3) & 1u;                 // K 8..15 / 24..31 -> hi lanes
    unsigned i        = (kk & 7u) + ((kk >> 4) << 3);   // elem slot within v16
    unsigned lane     = (r & 15u) + (lanehalf << 4);
    unsigned tile     = (r >> 4) * (Ktot >> 5) + (k >> 5);
    return tile * 512u + lane * 16u + i;
}

__device__ __forceinline__ float gelu_tanh(float x) {
    const float c = 0.7978845608028654f;   // sqrt(2/pi)
    return 0.5f * x * (1.0f + tanhf(c * (x + 0.044715f * x * x * x)));
}

// ---------------------------------------------------------------------------
// Weight cast + pack: fp32 row-major W[K][N]  ->  bf16 B-fragment layout
// ---------------------------------------------------------------------------
__global__ void pack_weight_kernel(const float* __restrict__ W,
                                   unsigned short* __restrict__ out,
                                   int K, int N) {
    int idx = blockIdx.x * blockDim.x + threadIdx.x;
    if (idx >= K * N) return;
    int k = idx / N, n = idx - k * N;
    out[fragIdx((unsigned)n, (unsigned)k, (unsigned)K)] = f2bf(W[idx]);
}

// ---------------------------------------------------------------------------
// Per-batch valid-token counts from the additive mask
// ---------------------------------------------------------------------------
__global__ void lens_kernel(const float* __restrict__ amask, int* __restrict__ lens) {
    __shared__ int red[256];
    int b = blockIdx.x, c = 0;
    for (int s = threadIdx.x; s < S_LEN; s += 256)
        c += (amask[b * S_LEN + s] >= -9000.0f) ? 1 : 0;
    red[threadIdx.x] = c; __syncthreads();
    for (int st = 128; st > 0; st >>= 1) {
        if (threadIdx.x < st) red[threadIdx.x] += red[threadIdx.x + st];
        __syncthreads();
    }
    if (threadIdx.x == 0) lens[b] = red[0];
}

// h[b,s,:] = registers[s%R] for s < S-len ; hidden[b,s] for S-len<=s<len ; 0 after
__global__ void build_h_kernel(const float* __restrict__ hs,
                               const float* __restrict__ regs,
                               const int* __restrict__ lens,
                               float* __restrict__ h) {
    int idx = blockIdx.x * blockDim.x + threadIdx.x;
    if (idx >= NB * S_LEN * DMODEL) return;
    int d = idx & (DMODEL - 1);
    int s = (idx >> 10) & (S_LEN - 1);
    int b = idx >> 20;
    int len = lens[b];
    float val;
    if (s >= S_LEN - len) val = (s < len) ? hs[idx] : 0.0f;
    else                  val = regs[(s & (NREG - 1)) * DMODEL + d];
    h[idx] = val;
}

// ---------------------------------------------------------------------------
// RMSNorm: one block per row; PACK=true -> bf16 A-fragment out, else fp32 out
// ---------------------------------------------------------------------------
template <bool PACK>
__global__ void rms_kernel(const float* __restrict__ x,
                           float* __restrict__ outF,
                           unsigned short* __restrict__ outP) {
    __shared__ float red[256];
    int row = blockIdx.x;
    const float* xr = x + (size_t)row * DMODEL;
    float ss = 0.0f;
    for (int d = threadIdx.x; d < DMODEL; d += 256) { float v = xr[d]; ss += v * v; }
    red[threadIdx.x] = ss; __syncthreads();
    for (int st = 128; st > 0; st >>= 1) {
        if (threadIdx.x < st) red[threadIdx.x] += red[threadIdx.x + st];
        __syncthreads();
    }
    float inv = rsqrtf(red[0] * (1.0f / DMODEL) + 1e-6f);
    for (int d = threadIdx.x; d < DMODEL; d += 256) {
        float v = xr[d] * inv;
        if (PACK) outP[fragIdx((unsigned)row, (unsigned)d, DMODEL)] = f2bf(v);
        else      outF[(size_t)row * DMODEL + d] = v;
    }
}

// ---------------------------------------------------------------------------
// Generic GEMM: A-frag(MxK) x B-frag(KxN) + bias [+ residual] -> fp32 row-major
// block = 8 waves; each wave computes a 32(M) x 64(N) tile  (block: 256M x 64N)
// per k-step: 2 A-frag + 4 B-frag loads (12 x b128) feed 8 WMMAs.
// ---------------------------------------------------------------------------
__global__ void gemm_kernel(const unsigned short* __restrict__ Ap,
                            const unsigned short* __restrict__ Bp,
                            const float* __restrict__ bias,
                            const float* __restrict__ residual,
                            float* __restrict__ out,
                            int M, int N, int K) {
    int lane = threadIdx.x & 31;
    int wave = threadIdx.x >> 5;
    int m0 = (blockIdx.y * 8 + wave) * 32;
    int n0 = blockIdx.x * 64;
    if (m0 >= M || n0 >= N) return;
    const int kt = K >> 5;
    const unsigned short* aBase = Ap + (size_t)(m0 >> 4) * kt * 512 + lane * 16;
    v8f acc[2][4] = {};
    for (int kc = 0; kc < kt; ++kc) {
        v16bf a0 = *reinterpret_cast<const v16bf*>(aBase + (size_t)kc * 512);
        v16bf a1 = *reinterpret_cast<const v16bf*>(aBase + (size_t)(kt + kc) * 512);
        const unsigned short* bBase = Bp + ((size_t)(n0 >> 4) * kt + kc) * 512 + lane * 16;
        __builtin_prefetch(aBase + (size_t)(kc + 2) * 512, 0, 3);
        __builtin_prefetch(aBase + (size_t)(kt + kc + 2) * 512, 0, 3);
#pragma unroll
        for (int j = 0; j < 4; ++j) {
            v16bf b = *reinterpret_cast<const v16bf*>(bBase + (size_t)j * kt * 512);
            acc[0][j] = WMMA_BF16(a0, b, acc[0][j]);
            acc[1][j] = WMMA_BF16(a1, b, acc[1][j]);
        }
    }
    int col = n0 + (lane & 15);
#pragma unroll
    for (int mi = 0; mi < 2; ++mi) {
        int rowB = m0 + mi * 16 + ((lane >> 4) << 3);
#pragma unroll
        for (int j = 0; j < 4; ++j) {
            int n = col + j * 16;
            float bv = bias ? bias[n] : 0.0f;
#pragma unroll
            for (int r = 0; r < 8; ++r) {
                size_t idx = (size_t)(rowB + r) * N + n;
                float v = acc[mi][j][r] + bv;
                if (residual) v += residual[idx];
                out[idx] = v;
            }
        }
    }
}

// ---------------------------------------------------------------------------
// Gated-MLP GEMM: u = n@W1 + b1 ; out = a * gelu(g)  packed to A-frag bf16
// W1 is K=1024 x N=8192; "a" cols [0,4096), "g" cols [4096,8192)
// wave tile: 32(M) x 64(N-pair): 16 WMMAs per k-step, 20 x b128 loads.
// ---------------------------------------------------------------------------
__global__ void gemm_gated_kernel(const unsigned short* __restrict__ Ap,
                                  const unsigned short* __restrict__ Bp,
                                  const float* __restrict__ bias,
                                  unsigned short* __restrict__ outP) {
    int lane = threadIdx.x & 31;
    int wave = threadIdx.x >> 5;
    int m0 = (blockIdx.y * 8 + wave) * 32;
    int n0 = blockIdx.x * 64;                 // within a-half [0,4096)
    const int kt = DMODEL >> 5;               // 32
    const unsigned short* aBase = Ap + (size_t)(m0 >> 4) * kt * 512 + lane * 16;
    v8f accA[2][4] = {}, accG[2][4] = {};
    for (int kc = 0; kc < kt; ++kc) {
        v16bf a0 = *reinterpret_cast<const v16bf*>(aBase + (size_t)kc * 512);
        v16bf a1 = *reinterpret_cast<const v16bf*>(aBase + (size_t)(kt + kc) * 512);
        __builtin_prefetch(aBase + (size_t)(kc + 2) * 512, 0, 3);
#pragma unroll
        for (int j = 0; j < 4; ++j) {
            const unsigned short* bA =
                Bp + ((size_t)((n0 + j * 16) >> 4) * kt + kc) * 512 + lane * 16;
            const unsigned short* bG =
                Bp + ((size_t)((n0 + 4096 + j * 16) >> 4) * kt + kc) * 512 + lane * 16;
            v16bf vbA = *reinterpret_cast<const v16bf*>(bA);
            v16bf vbG = *reinterpret_cast<const v16bf*>(bG);
            accA[0][j] = WMMA_BF16(a0, vbA, accA[0][j]);
            accA[1][j] = WMMA_BF16(a1, vbA, accA[1][j]);
            accG[0][j] = WMMA_BF16(a0, vbG, accG[0][j]);
            accG[1][j] = WMMA_BF16(a1, vbG, accG[1][j]);
        }
    }
    int col = n0 + (lane & 15);
#pragma unroll
    for (int mi = 0; mi < 2; ++mi) {
        int rowB = m0 + mi * 16 + ((lane >> 4) << 3);
#pragma unroll
        for (int j = 0; j < 4; ++j) {
            int n = col + j * 16;
            float ba = bias[n], bg = bias[n + 4096];
#pragma unroll
            for (int r = 0; r < 8; ++r) {
                float av = accA[mi][j][r] + ba;
                float gv = accG[mi][j][r] + bg;
                outP[fragIdx((unsigned)(rowB + r), (unsigned)n, 4096u)] =
                    f2bf(av * gelu_tanh(gv));
            }
        }
    }
}

// ---------------------------------------------------------------------------
// RoPE + repack into attention fragment layouts (bf16).
//   doRope=1: q or k  -> fragIdx(s, hd, 64)   (A-frag of K rows / B-frag of Q^T)
//   doRope=0: v       -> fragIdx(hd, s, 1024) (A-frag of V^T)
// one thread per (b, h, s, rotation pair)
// ---------------------------------------------------------------------------
__global__ void rope_pack_kernel(const float* __restrict__ xf,
                                 unsigned short* __restrict__ outP, int doRope) {
    int idx = blockIdx.x * blockDim.x + threadIdx.x;
    if (idx >= NB * NH * S_LEN * 32) return;
    int d2 = idx & 31;
    int t  = idx >> 5;
    int s  = t & (S_LEN - 1); t >>= 10;
    int hh = t & (NH - 1);
    int b  = t >> 4;
    size_t base = ((size_t)b * S_LEN + s) * DMODEL + hh * HDIM + 2 * d2;
    float x1 = xf[base], x2 = xf[base + 1];
    float o1, o2;
    if (doRope) {
        float freq = __expf(-(float)(2 * d2) * (9.210340371976184f / 64.0f)); // 1e4^(-2i/64)
        float sn, cs;
        __sincosf((float)s * freq, &sn, &cs);
        o1 = x1 * cs - x2 * sn;
        o2 = x1 * sn + x2 * cs;
    } else { o1 = x1; o2 = x2; }
    unsigned short* dst = outP + (size_t)(b * NH + hh) * (S_LEN * HDIM);
    if (doRope) {
        dst[fragIdx((unsigned)s, (unsigned)(2 * d2),     HDIM)] = f2bf(o1);
        dst[fragIdx((unsigned)s, (unsigned)(2 * d2 + 1), HDIM)] = f2bf(o2);
    } else {
        dst[fragIdx((unsigned)(2 * d2),     (unsigned)s, S_LEN)] = f2bf(o1);
        dst[fragIdx((unsigned)(2 * d2 + 1), (unsigned)s, S_LEN)] = f2bf(o2);
    }
}

// ---------------------------------------------------------------------------
// Flash attention, transposed-score trick. One wave per 16-query tile.
// S^T tile = K_tile(16x64) @ Q^T(64x16): its C-layout (per lane: 16 scores of
// ONE query) is byte-identical to the B-fragment of P^T over a 32-key chunk,
// so the softmax'd probs feed O^T += V^T @ P^T with no cross-lane traffic.
// ---------------------------------------------------------------------------
__global__ void attn_kernel(const unsigned short* __restrict__ qp,
                            const unsigned short* __restrict__ kp,
                            const unsigned short* __restrict__ vp,
                            unsigned short* __restrict__ op) {
    int lane = threadIdx.x & 31;
    int wave = threadIdx.x >> 5;
    int bh = blockIdx.x;                     // 0..63 = b*NH + h
    int qt = blockIdx.y * 4 + wave;          // 0..63 query tile
    const unsigned short* qb = qp + (size_t)bh * (S_LEN * HDIM);
    const unsigned short* kb = kp + (size_t)bh * (S_LEN * HDIM);
    const unsigned short* vb = vp + (size_t)bh * (S_LEN * HDIM);

    v16bf bq0 = *reinterpret_cast<const v16bf*>(qb + (size_t)(qt * 2 + 0) * 512 + lane * 16);
    v16bf bq1 = *reinterpret_cast<const v16bf*>(qb + (size_t)(qt * 2 + 1) * 512 + lane * 16);

    v8f o0 = {}, o1 = {}, o2 = {}, o3 = {};
    float mrun = -1e30f, lrun = 0.0f;
    const float scale = 0.125f;              // 1/sqrt(64)

    for (int kc = 0; kc < S_LEN / 32; ++kc) {
        v8f s0 = {}, s1 = {};
        {   // key tile 2*kc   (keys kc*32 .. +15)
            v16bf ak0 = *reinterpret_cast<const v16bf*>(kb + (size_t)((kc * 2 + 0) * 2 + 0) * 512 + lane * 16);
            v16bf ak1 = *reinterpret_cast<const v16bf*>(kb + (size_t)((kc * 2 + 0) * 2 + 1) * 512 + lane * 16);
            s0 = WMMA_BF16(ak0, bq0, s0);
            s0 = WMMA_BF16(ak1, bq1, s0);
        }
        {   // key tile 2*kc+1 (keys kc*32+16 .. +31)
            v16bf ak0 = *reinterpret_cast<const v16bf*>(kb + (size_t)((kc * 2 + 1) * 2 + 0) * 512 + lane * 16);
            v16bf ak1 = *reinterpret_cast<const v16bf*>(kb + (size_t)((kc * 2 + 1) * 2 + 1) * 512 + lane * 16);
            s1 = WMMA_BF16(ak0, bq0, s1);
            s1 = WMMA_BF16(ak1, bq1, s1);
        }
        // --- online softmax over this 32-key chunk (per-lane = one query) ---
        float mx = -1e30f;
#pragma unroll
        for (int r = 0; r < 8; ++r) {
            s0[r] *= scale; s1[r] *= scale;
            mx = fmaxf(mx, fmaxf(s0[r], s1[r]));
        }
        mx = fmaxf(mx, __shfl_xor(mx, 16, 32));
        float mnew = fmaxf(mrun, mx);
        float corr = __expf(mrun - mnew);
        union { unsigned short u[16]; v16bf v; } pb;
        float ls = 0.0f;
#pragma unroll
        for (int r = 0; r < 8; ++r) { float p = __expf(s0[r] - mnew); ls += p; pb.u[r]     = f2bf(p); }
#pragma unroll
        for (int r = 0; r < 8; ++r) { float p = __expf(s1[r] - mnew); ls += p; pb.u[8 + r] = f2bf(p); }
        ls += __shfl_xor(ls, 16, 32);
        lrun = lrun * corr + ls;
        mrun = mnew;
#pragma unroll
        for (int r = 0; r < 8; ++r) { o0[r] *= corr; o1[r] *= corr; o2[r] *= corr; o3[r] *= corr; }
        // --- O^T += V^T(hd tile) @ P^T ---
        o0 = WMMA_BF16(*reinterpret_cast<const v16bf*>(vb + (size_t)(0 * 32 + kc) * 512 + lane * 16), pb.v, o0);
        o1 = WMMA_BF16(*reinterpret_cast<const v16bf*>(vb + (size_t)(1 * 32 + kc) * 512 + lane * 16), pb.v, o1);
        o2 = WMMA_BF16(*reinterpret_cast<const v16bf*>(vb + (size_t)(2 * 32 + kc) * 512 + lane * 16), pb.v, o2);
        o3 = WMMA_BF16(*reinterpret_cast<const v16bf*>(vb + (size_t)(3 * 32 + kc) * 512 + lane * 16), pb.v, o3);
    }
    float invl = 1.0f / lrun;
    int b  = bh >> 4, hh = bh & 15;
    int m  = b * S_LEN + qt * 16 + (lane & 15);       // token row
    int hd0 = (lane >> 4) << 3;
    v8f oacc[4] = { o0, o1, o2, o3 };
#pragma unroll
    for (int j = 0; j < 4; ++j) {
#pragma unroll
        for (int r = 0; r < 8; ++r) {
            int kcol = hh * HDIM + j * 16 + hd0 + r;  // column in D
            op[fragIdx((unsigned)m, (unsigned)kcol, DMODEL)] = f2bf(oacc[j][r] * invl);
        }
    }
}

__global__ void zero_kernel(float* __restrict__ p, int n) {
    int i = blockIdx.x * blockDim.x + threadIdx.x;
    if (i < n) p[i] = 0.0f;
}

// ---------------------------------------------------------------------------
// Host orchestration
// ---------------------------------------------------------------------------
extern "C" void kernel_launch(void* const* d_in, const int* in_sizes, int n_in,
                              void* d_out, int out_size, void* d_ws, size_t ws_size,
                              hipStream_t stream) {
    (void)in_sizes; (void)n_in; (void)out_size; (void)ws_size;
    const float* hs    = (const float*)d_in[0];
    const float* amask = (const float*)d_in[1];
    const float* regs  = (const float*)d_in[2];
    const float* wq = (const float*)d_in[3];
    const float* bq = (const float*)d_in[4];
    const float* wk = (const float*)d_in[5];
    const float* bk = (const float*)d_in[6];
    const float* wv = (const float*)d_in[7];
    const float* bv = (const float*)d_in[8];
    const float* wo = (const float*)d_in[9];
    const float* bo = (const float*)d_in[10];
    const float* w1 = (const float*)d_in[11];
    const float* b1 = (const float*)d_in[12];
    const float* w2 = (const float*)d_in[13];
    const float* b2 = (const float*)d_in[14];
    float* out = (float*)d_out;

    char* ws = (char*)d_ws;
    size_t off = 0;
    auto take = [&](size_t bytes) -> char* {
        char* p = ws + off;
        off = (off + bytes + 255) & ~(size_t)255;
        return p;
    };
    const size_t DD  = (size_t)DMODEL * DMODEL;       // 1M
    const size_t D8  = (size_t)DMODEL * 8 * DMODEL;   // 8M
    const size_t D4  = (size_t)4 * DMODEL * DMODEL;   // 4M
    unsigned short* wqP = (unsigned short*)take(2 * NLAYER * DD);
    unsigned short* wkP = (unsigned short*)take(2 * NLAYER * DD);
    unsigned short* wvP = (unsigned short*)take(2 * NLAYER * DD);
    unsigned short* woP = (unsigned short*)take(2 * NLAYER * DD);
    unsigned short* w1P = (unsigned short*)take(2 * NLAYER * D8);
    unsigned short* w2P = (unsigned short*)take(2 * NLAYER * D4);
    float*          h   = (float*)take(4 * (size_t)MTOK * DMODEL);
    unsigned short* nP  = (unsigned short*)take(2 * (size_t)MTOK * DMODEL);
    float*          xf  = (float*)take(4 * (size_t)MTOK * DMODEL);
    unsigned short* qP  = (unsigned short*)take(2 * (size_t)NB * NH * S_LEN * HDIM);
    unsigned short* kP  = (unsigned short*)take(2 * (size_t)NB * NH * S_LEN * HDIM);
    unsigned short* vP  = (unsigned short*)take(2 * (size_t)NB * NH * S_LEN * HDIM);
    unsigned short* oP  = (unsigned short*)take(2 * (size_t)MTOK * DMODEL);
    unsigned short* mP  = (unsigned short*)take(2 * (size_t)MTOK * 4 * DMODEL);
    int*            lens = (int*)take(sizeof(int) * NB);

    // ---- cast + pack all weights to bf16 fragment layout ----
    for (int l = 0; l < NLAYER; ++l) {
        pack_weight_kernel<<<(int)((DD + 255) / 256), 256, 0, stream>>>(wq + l * DD, wqP + l * DD, DMODEL, DMODEL);
        pack_weight_kernel<<<(int)((DD + 255) / 256), 256, 0, stream>>>(wk + l * DD, wkP + l * DD, DMODEL, DMODEL);
        pack_weight_kernel<<<(int)((DD + 255) / 256), 256, 0, stream>>>(wv + l * DD, wvP + l * DD, DMODEL, DMODEL);
        pack_weight_kernel<<<(int)((DD + 255) / 256), 256, 0, stream>>>(wo + l * DD, woP + l * DD, DMODEL, DMODEL);
        pack_weight_kernel<<<(int)((D8 + 255) / 256), 256, 0, stream>>>(w1 + l * D8, w1P + l * D8, DMODEL, 8 * DMODEL);
        pack_weight_kernel<<<(int)((D4 + 255) / 256), 256, 0, stream>>>(w2 + l * D4, w2P + l * D4, 4 * DMODEL, DMODEL);
    }

    // ---- preprocess: token packing with registers ----
    lens_kernel<<<NB, 256, 0, stream>>>(amask, lens);
    build_h_kernel<<<(NB * S_LEN * DMODEL) / 256, 256, 0, stream>>>(hs, regs, lens, h);

    const dim3 gGemmD(DMODEL / 64, MTOK / 256);        // N=1024 GEMMs, 32-row waves
    const dim3 gGated(4096 / 64, MTOK / 256);
    const dim3 gAttn(NB * NH, S_LEN / 64);
    const int  nRope = NB * NH * S_LEN * 32;

    for (int l = 0; l < NLAYER; ++l) {
        // n = rms(h), packed
        rms_kernel<true><<<MTOK, 256, 0, stream>>>(h, nullptr, nP);
        // Q
        gemm_kernel<<<gGemmD, 256, 0, stream>>>(nP, wqP + l * DD, bq + l * DMODEL, nullptr, xf, MTOK, DMODEL, DMODEL);
        rope_pack_kernel<<<nRope / 256, 256, 0, stream>>>(xf, qP, 1);
        // K
        gemm_kernel<<<gGemmD, 256, 0, stream>>>(nP, wkP + l * DD, bk + l * DMODEL, nullptr, xf, MTOK, DMODEL, DMODEL);
        rope_pack_kernel<<<nRope / 256, 256, 0, stream>>>(xf, kP, 1);
        // V
        gemm_kernel<<<gGemmD, 256, 0, stream>>>(nP, wvP + l * DD, bv + l * DMODEL, nullptr, xf, MTOK, DMODEL, DMODEL);
        rope_pack_kernel<<<nRope / 256, 256, 0, stream>>>(xf, vP, 0);
        // attention -> oP (A-frag layout)
        attn_kernel<<<gAttn, 128, 0, stream>>>(qP, kP, vP, oP);
        // h = o @ wo + bo + h
        gemm_kernel<<<gGemmD, 256, 0, stream>>>(oP, woP + l * DD, bo + l * DMODEL, h, h, MTOK, DMODEL, DMODEL);
        // n = rms(h), packed
        rms_kernel<true><<<MTOK, 256, 0, stream>>>(h, nullptr, nP);
        // m = a * gelu(g), packed
        gemm_gated_kernel<<<gGated, 256, 0, stream>>>(nP, w1P + l * D8, b1 + l * 8 * DMODEL, mP);
        // h = m @ w2 + b2 + h
        gemm_kernel<<<gGemmD, 256, 0, stream>>>(mP, w2P + l * D4, b2 + l * DMODEL, h, h, MTOK, DMODEL, 4 * DMODEL);
    }

    // ---- outputs: rms(h) fp32, then 4096 zeros for amask ----
    rms_kernel<false><<<MTOK, 256, 0, stream>>>(h, out, nullptr);
    zero_kernel<<<(NB * S_LEN + 255) / 256, 256, 0, stream>>>(out + (size_t)MTOK * DMODEL, NB * S_LEN);
}
